// ConvolutionalLayer_1P_V2_24507083391347
// MI455X (gfx1250) — matrix operand order repaired and checked by reading, not compiled, not run on hardware
//
#include <hip/hip_runtime.h>

// ---------------------------------------------------------------------------
// Problem constants (from reference): N=50000, C=256, D=25000, K=16, C_OUT=256
// Y[400000, 256] = G[400000,256] @ Weff[256,256] + b
//   G[r,c] = x[domains[r], c] * symm_norm[domains[r]]
//   Weff   = w[0:256,:] + w[256:512,:]
//
// Roofline: 52.4 GFLOP; ~464 MB HBM traffic -> ~20 us floor @ 23.3 TB/s.
// bf16 hi/lo split (3x v_wmma_f32_16x16x32_bf16, f32 accum) gives ~2^-17
// relative error at ~8x the per-instruction K-throughput of fp32 WMMA.
// ---------------------------------------------------------------------------

typedef __attribute__((ext_vector_type(4)))  float  f32x4;
typedef __attribute__((ext_vector_type(8)))  float  v8f;
typedef __attribute__((ext_vector_type(16))) __bf16 v16bf;

#define C_DIM   256
#define ROWS    400000      // D*K = 25000*16
#define TILE_R  64          // rows per workgroup

// ---------------------------------------------------------------------------
// Prep: Weff = w[k][n] + w[k+256][n], split into bf16 hi/lo, stored transposed
// as Wt[n][k] so B-fragments are contiguous 32B per lane.
// grid = 256 blocks (n), block = 256 threads (k)
// ---------------------------------------------------------------------------
__global__ __launch_bounds__(256) void prep_w_kernel(
    const float* __restrict__ w, __bf16* __restrict__ whi,
    __bf16* __restrict__ wlo)
{
    const int n = blockIdx.x;
    const int k = threadIdx.x;
    const float v  = w[k * C_DIM + n] + w[(k + C_DIM) * C_DIM + n];
    const __bf16 h = (__bf16)v;               // RNE
    const __bf16 l = (__bf16)(v - (float)h);  // residual
    whi[n * C_DIM + k] = h;
    wlo[n * C_DIM + k] = l;
}

// ---------------------------------------------------------------------------
// Gather-GEMM with bf16 hi/lo split, f32 accumulation via WMMA.
// Block = 256 threads = 8 waves (wave32). Tile: 64 rows x 256 cols.
// Wave (rg, cg): rg in 0..3 -> 16 rows, cg in 0..1 -> 128 cols (8 fragments).
// No LDS, no barriers: A gathered from L2-resident x, B from ws (256 KB, L2).
//
// K-loop unrolled by 2 (4 trips): ~180 live VGPRs -> fits the 256-VGPR cap
// from __launch_bounds__(256, 4) with NO scratch spills and no
// s_set_vgpr_msb traffic, while exposing two independent K-chunks of
// WMMA + load ILP per iteration.
// ---------------------------------------------------------------------------
__global__ __launch_bounds__(256, 4) void gather_gemm_kernel(
    const float* __restrict__ x,          // [50000, 256]
    const float* __restrict__ symm_norm,  // [50000]
    const int*   __restrict__ domains,    // [400000] flat (row-major D x K)
    const __bf16* __restrict__ whi,       // [256 n][256 k]
    const __bf16* __restrict__ wlo,       // [256 n][256 k]
    const float* __restrict__ bias,       // [256]
    float*       __restrict__ out)        // [400000, 256]
{
    const int tid  = threadIdx.x;
    const int lane = tid & 31;
    const int wave = tid >> 5;
    const int cg   = wave & 1;    // column group: 128 cols each
    const int rg   = wave >> 1;   // row group: 16 rows each
    const int hlf  = lane >> 4;   // lane half (ISA A/B layout)
    const int l15  = lane & 15;

    const int rbase = blockIdx.x * TILE_R + rg * 16;
    // A-fragment row for this lane (M = lane&15, both halves same row)
    const int r    = rbase + l15;
    const int src  = domains[r];
    const float s  = symm_norm[src];
    const float* __restrict__ px = x + (size_t)src * C_DIM;

    v8f acc[8];
#pragma unroll
    for (int j = 0; j < 8; ++j)
#pragma unroll
        for (int i = 0; i < 8; ++i)
            acc[j][i] = 0.0f;

#pragma unroll 2
    for (int kc = 0; kc < 8; ++kc) {
        const int kbase = kc * 32;

        // --- A fragment (16x32 bf16). Layout: element e -> K = 16*hlf + e
        // (e=0..15): one contiguous 64B run (matches B).
        const float* pa = px + kbase + 16 * hlf;
        const f32x4* pq = reinterpret_cast<const f32x4*>(pa);
        f32x4 q[4];
        q[0] = pq[0];  // K: 16h + 0..3
        q[1] = pq[1];  // K: 16h + 4..7
        q[2] = pq[2];  // K: 16h + 8..11
        q[3] = pq[3];  // K: 16h + 12..15

        v16bf ahi, alo;
#pragma unroll
        for (int t = 0; t < 4; ++t)
#pragma unroll
            for (int e = 0; e < 4; ++e) {
                const float v  = q[t][e] * s;
                const __bf16 h = (__bf16)v;
                ahi[t * 4 + e] = h;
                alo[t * 4 + e] = (__bf16)(v - (float)h);
            }

        // --- 8 column fragments, 3 WMMA products each (hi*hi + lo*hi + hi*lo)
#pragma unroll
        for (int j = 0; j < 8; ++j) {
            const int n = cg * 128 + j * 16 + l15;
            // B layout: lanes 0-15 K=kbase..+15, lanes 16-31 K=kbase+16..+31,
            // sequential per lane -> one contiguous 32B read of Wt[n][k...]
            const size_t boff = (size_t)n * C_DIM + kbase + 16 * hlf;
            const v16bf bh = *reinterpret_cast<const v16bf*>(whi + boff);
            const v16bf bl = *reinterpret_cast<const v16bf*>(wlo + boff);

            acc[j] = __builtin_amdgcn_wmma_f32_16x16x32_bf16(
                false, ahi, false, bh, (short)0, acc[j], false, false);
            acc[j] = __builtin_amdgcn_wmma_f32_16x16x32_bf16(
                false, alo, false, bh, (short)0, acc[j], false, false);
            acc[j] = __builtin_amdgcn_wmma_f32_16x16x32_bf16(
                false, ahi, false, bl, (short)0, acc[j], false, false);
        }
    }

    // --- Epilogue: bias add + non-temporal store (keep x/W resident in L2).
    // D layout: element i -> row M = i + 8*hlf, col N = lane&15.
#pragma unroll
    for (int j = 0; j < 8; ++j) {
        const int n   = cg * 128 + j * 16 + l15;
        const float bv = bias[n];
#pragma unroll
        for (int i = 0; i < 8; ++i) {
            const int row = rbase + i + 8 * hlf;
            __builtin_nontemporal_store(acc[j][i] + bv,
                                        out + (size_t)row * C_DIM + n);
        }
    }
}

// ---------------------------------------------------------------------------
// Inputs (setup_inputs order): x, symm_norm, domains, w, b
// ---------------------------------------------------------------------------
extern "C" void kernel_launch(void* const* d_in, const int* in_sizes, int n_in,
                              void* d_out, int out_size, void* d_ws, size_t ws_size,
                              hipStream_t stream) {
    const float* x         = (const float*)d_in[0];
    const float* symm_norm = (const float*)d_in[1];
    const int*   domains   = (const int*)  d_in[2];
    const float* w         = (const float*)d_in[3];
    const float* bias      = (const float*)d_in[4];
    float*       out       = (float*)d_out;

    // ws: Wt_hi [256*256 bf16] then Wt_lo [256*256 bf16] = 256 KB total
    __bf16* whi = (__bf16*)d_ws;
    __bf16* wlo = whi + C_DIM * C_DIM;

    prep_w_kernel<<<C_DIM, C_DIM, 0, stream>>>(w, whi, wlo);

    gather_gemm_kernel<<<ROWS / TILE_R, 256, 0, stream>>>(
        x, symm_norm, domains, whi, wlo, bias, out);
}
